// SetAbstraction_65120294142109
// MI455X (gfx1250) — matrix operand order repaired and checked by reading, not compiled, not run on hardware
//
#include <hip/hip_runtime.h>
#include <hip/hip_bf16.h>

typedef __attribute__((ext_vector_type(16))) _Float16 v16h;
typedef __attribute__((ext_vector_type(8)))  _Float16 v8h;
typedef __attribute__((ext_vector_type(8)))  float    v8f;

#define B_   16
#define N_   4096
#define M_   1024
#define K_   32
#define C_   64
#define ROWS (B_ * M_ * K_)        // 524288 grouped points
#define R2   0.04f
#define EPS_ 1e-5f

// ---------------------------------------------------------------- utilities
__global__ void zero_stats_kernel(float* stats) {
    int i = blockIdx.x * blockDim.x + threadIdx.x;
    if (i < 512) stats[i] = 0.f;
}

// Pack W[COUT x wstride] (f32, row-major) into the WMMA B-fragment layout:
// Wp[((colT*KS + ks)*32 + lane)*16 + s] = W[(colT*16 + lane%16)*wstride + ks*32 + k], k = 8*(lane/16) + s%8 + 16*(s/8) (0 if >= kmax)
__global__ void pack_w_kernel(const float* __restrict__ W, _Float16* __restrict__ Wp,
                              int wstride, int kmax, int CT, int KS) {
    int i = blockIdx.x * 256 + threadIdx.x;
    if (i >= CT * KS * 512) return;
    int n    = (i >> 4) & 15;
    int k    = 8 * ((i >> 8) & 1) + (i & 7) + 16 * ((i >> 3) & 1);
    int ks   = (i >> 9) % KS;
    int colT = i / (512 * KS);
    int krow = ks * 32 + k;
    float w = (krow < kmax) ? W[(size_t)(colT * 16 + n) * wstride + krow] : 0.f;
    Wp[i] = (_Float16)w;
}

// ---------------------------------------------------------------- 1) FPS
// One block per batch. 1024 threads, each owns 4 points (n = t + i*1024).
__global__ __launch_bounds__(1024)
void fps_kernel(const float* __restrict__ xyz, float* __restrict__ out_xyz) {
    __shared__ unsigned long long red[1024];
    __shared__ float sc[3];
    const int b = blockIdx.x;
    const int t = threadIdx.x;

    float px[4], py[4], pz[4], pd[4];
#pragma unroll
    for (int i = 0; i < 4; ++i) {
        int n = t + i * 1024;
        const float* p = xyz + ((size_t)b * N_ + n) * 3;
        px[i] = p[0]; py[i] = p[1]; pz[i] = p[2];
        pd[i] = 1e10f;
    }

    int far = 0;
    for (int it = 0; it < M_; ++it) {
        if (t == 0) {
            const float* p = xyz + ((size_t)b * N_ + far) * 3;
            float x = p[0], y = p[1], z = p[2];
            sc[0] = x; sc[1] = y; sc[2] = z;
            float* o = out_xyz + ((size_t)b * M_ + it) * 3;
            o[0] = x; o[1] = y; o[2] = z;
        }
        __syncthreads();
        float cx = sc[0], cy = sc[1], cz = sc[2];

        unsigned long long key = 0;
#pragma unroll
        for (int i = 0; i < 4; ++i) {
            float dx = px[i] - cx, dy = py[i] - cy, dz = pz[i] - cz;
            float d2 = dx * dx + dy * dy + dz * dz;
            pd[i] = fminf(pd[i], d2);
            unsigned n = (unsigned)(t + i * 1024);
            unsigned long long k =
                ((unsigned long long)__float_as_uint(pd[i]) << 32) |
                (unsigned long long)(~n);
            if (k > key) key = k;
        }
        red[t] = key;
        __syncthreads();
        for (int s = 512; s > 0; s >>= 1) {
            if (t < s) {
                unsigned long long o = red[t + s];
                if (o > red[t]) red[t] = o;
            }
            __syncthreads();
        }
        far = (int)(~(unsigned)(red[0] & 0xffffffffu));
        __syncthreads();
    }
}

// ------------------------------------------------- 2) ball query + grouping
__global__ __launch_bounds__(256)
void group_kernel(const float* __restrict__ xyz, const float* __restrict__ pts,
                  const float* __restrict__ new_xyz, _Float16* __restrict__ X0) {
    __shared__ int sel[8][32];
    const int wv   = threadIdx.x >> 5;
    const int lane = threadIdx.x & 31;
    const int bm   = blockIdx.x * 8 + wv;
    const int b    = bm >> 10;

    sel[wv][lane] = 0;                       // default: point 0 (ref fallback)

    const float cx = new_xyz[(size_t)bm * 3 + 0];
    const float cy = new_xyz[(size_t)bm * 3 + 1];
    const float cz = new_xyz[(size_t)bm * 3 + 2];

    int count = 0;
    for (int chunk = 0; chunk < N_ / 32 && count < K_; ++chunk) {
        int n = chunk * 32 + lane;
        const float* p = xyz + ((size_t)b * N_ + n) * 3;
        float dx = p[0] - cx, dy = p[1] - cy, dz = p[2] - cz;
        bool in = (dx * dx + dy * dy + dz * dz) <= R2;
        unsigned mask = (unsigned)__ballot(in);
        int pos = count + __popc(mask & ((1u << lane) - 1u));
        if (in && pos < K_) sel[wv][pos] = n;
        count += __popc(mask);
    }
    __syncthreads();

    const int idx = sel[wv][lane];
    const float* p = xyz + ((size_t)b * N_ + idx) * 3;

    alignas(16) _Float16 tmp[96];
    tmp[0] = (_Float16)(p[0] - cx);
    tmp[1] = (_Float16)(p[1] - cy);
    tmp[2] = (_Float16)(p[2] - cz);
    const float4* f4 = (const float4*)(pts + ((size_t)b * N_ + idx) * (size_t)C_);
#pragma unroll
    for (int q = 0; q < 16; ++q) {
        float4 v = f4[q];
        tmp[3 + 4 * q + 0] = (_Float16)v.x;
        tmp[3 + 4 * q + 1] = (_Float16)v.y;
        tmp[3 + 4 * q + 2] = (_Float16)v.z;
        tmp[3 + 4 * q + 3] = (_Float16)v.w;
    }
#pragma unroll
    for (int c = 67; c < 96; ++c) tmp[c] = (_Float16)0.f;

    v8h* dst = (v8h*)(X0 + ((size_t)bm * K_ + lane) * 96);
#pragma unroll
    for (int j = 0; j < 12; ++j) dst[j] = *(const v8h*)(tmp + 8 * j);
}

// --------------------------------------------- 3) WMMA GEMM + bias + stats
// One wave owns a 16-row stripe: A fragments (KS x v16h) held in registers and
// reused across all CT column tiles; packed B served from LDS.
template <int CIN, int COUT, bool ACT>
__global__ __launch_bounds__(256)
void gemm_bn_kernel(const _Float16* __restrict__ X,
                    const _Float16* __restrict__ Wp,
                    const float* __restrict__ bias,
                    const float* __restrict__ sg, const float* __restrict__ tg,
                    _Float16* __restrict__ Y,
                    float* __restrict__ sumv, float* __restrict__ sumsq) {
    constexpr int CT = COUT / 16;
    constexpr int KS = CIN / 32;
    constexpr int WN = CT * KS * 512;          // packed B element count

    __shared__ _Float16 wlds[WN];
    __shared__ float ss[ACT ? CIN : 1];
    __shared__ float st[ACT ? CIN : 1];

    for (int i = threadIdx.x * 8; i < WN; i += 256 * 8)
        *(v8h*)(wlds + i) = *(const v8h*)(Wp + i);
    if (ACT) {
        for (int i = threadIdx.x; i < CIN; i += 256) { ss[i] = sg[i]; st[i] = tg[i]; }
    }
    __syncthreads();

    const int wv     = threadIdx.x >> 5;
    const int lane   = threadIdx.x & 31;
    const int stripe = blockIdx.x * 8 + wv;    // 16-row stripe id
    const int half   = lane >> 4;
    const int mlane  = lane & 15;

    const _Float16* xr = X + ((size_t)stripe * 16 + mlane) * CIN;

    // Build A fragments once; reuse for every column tile.
    v16h afrag[KS];
#pragma unroll
    for (int ks = 0; ks < KS; ++ks) {
        const v8h* ap = (const v8h*)(xr + ks * 32 + 8 * half);
        v8h alo = ap[0];          // K = ks*32 + 8*half + [0..7]
        v8h ahi = ap[2];          // K = ks*32 + 16 + 8*half + [0..7]
        if (ACT) {
#pragma unroll
            for (int e = 0; e < 8; ++e) {
                int c0 = ks * 32 + 8 * half + e;
                int c1 = c0 + 16;
                alo[e] = (_Float16)fmaxf(0.f, (float)alo[e] * ss[c0] + st[c0]);
                ahi[e] = (_Float16)fmaxf(0.f, (float)ahi[e] * ss[c1] + st[c1]);
            }
        }
        afrag[ks] = __builtin_shufflevector(alo, ahi,
                    0, 1, 2, 3, 4, 5, 6, 7, 8, 9, 10, 11, 12, 13, 14, 15);
    }

#pragma unroll
    for (int colT = 0; colT < CT; ++colT) {
        v8f acc = (v8f){0.f, 0.f, 0.f, 0.f, 0.f, 0.f, 0.f, 0.f};
#pragma unroll
        for (int ks = 0; ks < KS; ++ks) {
            const v8h* bp = (const v8h*)(wlds + ((colT * KS + ks) * 32 + lane) * 16);
            v16h b = __builtin_shufflevector(bp[0], bp[1],
                     0, 1, 2, 3, 4, 5, 6, 7, 8, 9, 10, 11, 12, 13, 14, 15);
            acc = __builtin_amdgcn_wmma_f32_16x16x32_f16(
                false, afrag[ks], false, b, (short)0, acc, false, false);
        }

        // Epilogue: VGPR r holds M=r (lanes<16) / M=8+r (lanes>=16), N=lane&15
        const int ch = colT * 16 + mlane;
        const float bch = bias[ch];
        float ps = 0.f, pq = 0.f;
#pragma unroll
        for (int r = 0; r < 8; ++r) {
            int mm = (lane < 16) ? r : (8 + r);
            size_t row = (size_t)stripe * 16 + mm;
            float y = acc[r] + bch;
            Y[row * COUT + ch] = (_Float16)y;
            ps += y;
            pq += y * y;
        }
        ps += __shfl_xor(ps, 16, 32);   // combine half-waves (same channel)
        pq += __shfl_xor(pq, 16, 32);
        if (lane < 16) {
            atomicAdd(sumv + ch, ps);
            atomicAdd(sumsq + ch, pq);
        }
    }
}

// ----------------------------------------- 4) BN scale/shift from batch stats
__global__ void scale_kernel(const float* __restrict__ sum, const float* __restrict__ sq,
                             const float* __restrict__ g, const float* __restrict__ be,
                             float* __restrict__ s, float* __restrict__ t, int C) {
    int c = threadIdx.x;
    if (c >= C) return;
    const float cnt  = (float)ROWS;
    float mean = sum[c] / cnt;
    float var  = sq[c] / cnt - mean * mean;
    float sc   = g[c] * rsqrtf(var + EPS_);
    s[c] = sc;
    t[c] = be[c] - mean * sc;
}

// -------------------------- 5) final BN+ReLU of layer 2 fused with max over K
__global__ __launch_bounds__(128)
void finalize_kernel(const _Float16* __restrict__ Y3,
                     const float* __restrict__ s, const float* __restrict__ t,
                     float* __restrict__ outp) {
    const int bm = blockIdx.x;
    const int c  = threadIdx.x;          // 0..127
    const float sc = s[c], tc = t[c];
    float mx = 0.f;                      // relu output lower bound
#pragma unroll 8
    for (int k = 0; k < K_; ++k) {
        float v = (float)Y3[((size_t)bm * K_ + k) * 128 + c] * sc + tc;
        mx = fmaxf(mx, fmaxf(v, 0.f));
    }
    outp[(size_t)bm * 128 + c] = mx;
}

// ---------------------------------------------------------------------------
extern "C" void kernel_launch(void* const* d_in, const int* in_sizes, int n_in,
                              void* d_out, int out_size, void* d_ws, size_t ws_size,
                              hipStream_t stream) {
    const float* xyz = (const float*)d_in[0];
    const float* pts = (const float*)d_in[1];
    const float* w0  = (const float*)d_in[2];
    const float* b0  = (const float*)d_in[3];
    const float* g0  = (const float*)d_in[4];
    const float* be0 = (const float*)d_in[5];
    const float* w1  = (const float*)d_in[6];
    const float* b1  = (const float*)d_in[7];
    const float* g1  = (const float*)d_in[8];
    const float* be1 = (const float*)d_in[9];
    const float* w2  = (const float*)d_in[10];
    const float* b2  = (const float*)d_in[11];
    const float* g2  = (const float*)d_in[12];
    const float* be2 = (const float*)d_in[13];

    float* out      = (float*)d_out;
    float* out_xyz  = out;                               // [B,M,3]
    float* out_pts  = out + (size_t)B_ * M_ * 3;         // [B,M,128]

    char* w = (char*)d_ws;
    _Float16* X0 = (_Float16*)(w);                       // 524288*96*2
    _Float16* Y1 = (_Float16*)(w + 100663296ull);        // 524288*64*2
    _Float16* Y2 = (_Float16*)(w + 167772160ull);        // 524288*64*2
    _Float16* Y3 = (_Float16*)(w + 234881024ull);        // 524288*128*2
    float* stats = (float*)(w + 369098752ull);
    float* sum0 = stats +   0, * sq0 = stats +  64;
    float* sum1 = stats + 128, * sq1 = stats + 192;
    float* sum2 = stats + 256, * sq2 = stats + 384;      // 128 each
    float* s0 = stats + 512, * t0 = stats + 576;
    float* s1 = stats + 640, * t1 = stats + 704;
    float* s2 = stats + 768, * t2 = stats + 896;         // 128 each
    _Float16* Wp0 = (_Float16*)(w + 369106944ull);       // 4*3*512 f16 = 12 KB
    _Float16* Wp1 = (_Float16*)(w + 369119232ull);       // 4*2*512 f16 =  8 KB
    _Float16* Wp2 = (_Float16*)(w + 369127424ull);       // 8*2*512 f16 = 16 KB

    zero_stats_kernel<<<1, 512, 0, stream>>>(stats);
    pack_w_kernel<<<24, 256, 0, stream>>>(w0, Wp0, 67, 67, 4, 3);
    pack_w_kernel<<<16, 256, 0, stream>>>(w1, Wp1, 64, 64, 4, 2);
    pack_w_kernel<<<32, 256, 0, stream>>>(w2, Wp2, 64, 64, 8, 2);

    fps_kernel<<<B_, 1024, 0, stream>>>(xyz, out_xyz);
    group_kernel<<<(B_ * M_) / 8, 256, 0, stream>>>(xyz, pts, out_xyz, X0);

    // One wave per 16-row stripe -> 32768 waves -> 4096 blocks per layer.
    gemm_bn_kernel<96, 64, false><<<ROWS / 16 / 8, 256, 0, stream>>>(
        X0, Wp0, b0, nullptr, nullptr, Y1, sum0, sq0);
    scale_kernel<<<1, 64, 0, stream>>>(sum0, sq0, g0, be0, s0, t0, 64);

    gemm_bn_kernel<64, 64, true><<<ROWS / 16 / 8, 256, 0, stream>>>(
        Y1, Wp1, b1, s0, t0, Y2, sum1, sq1);
    scale_kernel<<<1, 64, 0, stream>>>(sum1, sq1, g1, be1, s1, t1, 64);

    gemm_bn_kernel<64, 128, true><<<ROWS / 16 / 8, 256, 0, stream>>>(
        Y2, Wp2, b2, s1, t1, Y3, sum2, sq2);
    scale_kernel<<<1, 128, 0, stream>>>(sum2, sq2, g2, be2, s2, t2, 128);

    // BN2+ReLU fused with max-pool over K
    finalize_kernel<<<B_ * M_, 128, 0, stream>>>(Y3, s2, t2, out_pts);
}
